// CausalSelfAttention_90486370992354
// MI455X (gfx1250) — compile-verified, matrix operand order checked by reading, not run on hardware
//
#include <hip/hip_runtime.h>
#include <cstdint>

// ---------------------------------------------------------------------------
// Causal self-attention, B=2 S=2048 D=2048 H=16 HD=128, for MI455X (gfx1250).
// Compute-bound on WMMA (~206 GFLOP vs ~130MB HBM traffic): all three matmul
// stages run through v_wmma_f32_16x16x32_f16 (f32 accumulate). GEMMs use
// double-buffered LDS filled by GLOBAL_LOAD_ASYNC_TO_LDS_B128 (ASYNCcnt).
// ---------------------------------------------------------------------------

typedef __attribute__((ext_vector_type(16))) _Float16 v16h;
typedef __attribute__((ext_vector_type(8)))  _Float16 v8h;
typedef __attribute__((ext_vector_type(4)))  _Float16 v4h;
typedef __attribute__((ext_vector_type(8)))  float    f32x8;

#define S_LEN 2048
#define DMODEL 2048
#define NHEAD 16
#define HDIM 128
#define BS 2
// 1/sqrt(128) * log2(e): softmax computed in base 2 (v_exp_f32 is native exp2)
#define QSCALE_L2E (0.08838834764831845f * 1.44269504088896340f)

__device__ __forceinline__ f32x8 wmma16(v16h a, v16h b, f32x8 c) {
  return __builtin_amdgcn_wmma_f32_16x16x32_f16(
      /*neg_a=*/false, a, /*neg_b=*/false, b,
      /*c_mod=*/(short)0, c, /*reuse_a=*/false, /*reuse_b=*/false);
}

// Load one 16x32 A-layout (or B^T-layout) fragment from a row pointer.
// Per ISA: element i (i<8)  -> k = half*8 + i
//          element i (i>=8) -> k = 16 + half*8 + (i-8)
__device__ __forceinline__ v16h frag16(const _Float16* __restrict__ p, int half) {
  v8h lo = *(const v8h*)(p + half * 8);
  v8h hi = *(const v8h*)(p + 16 + half * 8);
  v16h r;
#pragma unroll
  for (int i = 0; i < 8; i++) { r[i] = lo[i]; r[i + 8] = hi[i]; }
  return r;
}

// Async global->LDS copy, 16B per lane (ISA 15.18.3 op 98). lds = wave-relative
// LDS byte offset (low 32 bits of generic pointer), tracked by ASYNCcnt.
__device__ __forceinline__ void async_b128(unsigned lds, const _Float16* gptr) {
  asm volatile("global_load_async_to_lds_b128 %0, %1, off"
               :: "v"(lds), "v"(gptr) : "memory");
}
__device__ __forceinline__ void wait_async0() {
  asm volatile("s_wait_asynccnt 0" ::: "memory");
}

// ---------------------------------------------------------------------------
// fp32 -> fp16 cast (x activations)
// ---------------------------------------------------------------------------
__global__ __launch_bounds__(256) void cast_f32_f16(const float* __restrict__ in,
                                                    _Float16* __restrict__ out) {
  int i = (blockIdx.x * 256 + threadIdx.x) * 4;
  float4 v = *(const float4*)(in + i);
  v4h o;
  o[0] = (_Float16)v.x; o[1] = (_Float16)v.y;
  o[2] = (_Float16)v.z; o[3] = (_Float16)v.w;
  *(v4h*)(out + i) = o;
}

// ---------------------------------------------------------------------------
// W[K][N] fp32  ->  W^T[N][K] fp16   (so GEMM B-fragments load contiguously)
// ---------------------------------------------------------------------------
__global__ __launch_bounds__(256) void transpose_cast(const float* __restrict__ in,
                                                      _Float16* __restrict__ out,
                                                      int Kd, int Nd) {
  __shared__ _Float16 t[32][33];
  int tx = threadIdx.x & 31, ty = threadIdx.x >> 5;      // ty: 0..7
  int n0 = blockIdx.x * 32, k0 = blockIdx.y * 32;
#pragma unroll
  for (int i = 0; i < 4; i++) {
    int kk = ty + i * 8;
    t[kk][tx] = (_Float16)in[(size_t)(k0 + kk) * Nd + n0 + tx];
  }
  __syncthreads();
#pragma unroll
  for (int i = 0; i < 4; i++) {
    int nn = ty + i * 8;
    out[(size_t)(n0 + nn) * Kd + k0 + tx] = t[tx][nn];
  }
}

// ---------------------------------------------------------------------------
// Tiled WMMA GEMM:  C[M][N] = A[M][K] * Bt[N][K]^T + bias
//   block tile 128x128, BK=32, 8 waves each computing a 32x64 sub-tile.
//   Double-buffered LDS, filled with async global->LDS b128 copies so the
//   copy of chunk i+1 overlaps the 8 WMMAs of chunk i.
//   MODE 0: QKV epilogue — scatter Q (pre-scaled), K [BH][S][HD],
//           V transposed [BH][HD][S], all fp16.
//   MODE 1: proj epilogue — fp32 output with bias.
// ---------------------------------------------------------------------------
#define LSTR 40   // LDS row stride in fp16 elems (80B = 5*16B: aligned, conflict-free)
#define BUFB (128 * LSTR * 2)   // bytes per stage buffer (10240)

template <int MODE>
__global__ __launch_bounds__(256) void gemm_f16(
    const _Float16* __restrict__ A, const _Float16* __restrict__ Bt,
    const float* __restrict__ bias, float* __restrict__ Cout,
    _Float16* __restrict__ Qo, _Float16* __restrict__ Ko, _Float16* __restrict__ Vto,
    int Kdim, int Ndim) {
  __shared__ __attribute__((aligned(16))) _Float16 As[2][128 * LSTR];
  __shared__ __attribute__((aligned(16))) _Float16 Bs[2][128 * LSTR];

  const int tid  = threadIdx.x;
  const int lane = tid & 31;
  const int wave = tid >> 5;
  const int wm   = wave >> 1;          // 0..3 : 32-row strip
  const int wn   = wave & 1;           // 0..1 : 64-col strip
  const int l16  = lane & 15;
  const int half = lane >> 4;
  const int m0   = blockIdx.y * 128;
  const int n0   = blockIdx.x * 128;

  // cooperative-copy coordinates: 2 threads per row, 16 elems (2 x b128) each
  const int lrow = tid >> 1;
  const int lcol = (tid & 1) * 16;
  const _Float16* gA = A  + (size_t)(m0 + lrow) * Kdim + lcol;
  const _Float16* gB = Bt + (size_t)(n0 + lrow) * Kdim + lcol;
  const unsigned ldsA = (unsigned)(uintptr_t)&As[0][lrow * LSTR + lcol];
  const unsigned ldsB = (unsigned)(uintptr_t)&Bs[0][lrow * LSTR + lcol];

  f32x8 acc[2][4] = {};
  const int nchunks = Kdim >> 5;

  // prime the pipeline: chunk 0 -> buffer 0
  async_b128(ldsA,      gA);
  async_b128(ldsA + 16, gA + 8);
  async_b128(ldsB,      gB);
  async_b128(ldsB + 16, gB + 8);

  for (int i = 0; i < nchunks; i++) {
    const int b = i & 1;
    wait_async0();        // my async writes have landed
    __syncthreads();      // everyone's writes landed; prior-buffer reads done

    if (i + 1 < nchunks) {           // prefetch chunk i+1 into buffer b^1
      const _Float16* ga = gA + (size_t)(i + 1) * 32;
      const _Float16* gb = gB + (size_t)(i + 1) * 32;
      const unsigned off = (unsigned)((b ^ 1) * BUFB);
      async_b128(ldsA + off,      ga);
      async_b128(ldsA + off + 16, ga + 8);
      async_b128(ldsB + off,      gb);
      async_b128(ldsB + off + 16, gb + 8);
    }

    v16h af[2], bfr[4];
#pragma unroll
    for (int tm = 0; tm < 2; tm++)
      af[tm] = frag16(&As[b][(wm * 32 + tm * 16 + l16) * LSTR], half);
#pragma unroll
    for (int tn = 0; tn < 4; tn++)
      bfr[tn] = frag16(&Bs[b][(wn * 64 + tn * 16 + l16) * LSTR], half);
#pragma unroll
    for (int tm = 0; tm < 2; tm++)
#pragma unroll
      for (int tn = 0; tn < 4; tn++)
        acc[tm][tn] = wmma16(af[tm], bfr[tn], acc[tm][tn]);
  }

  // Epilogue.  C layout: lane -> col n=l16, VGPR r -> row m = r + 8*half.
#pragma unroll
  for (int tm = 0; tm < 2; tm++) {
#pragma unroll
    for (int tn = 0; tn < 4; tn++) {
      int n = n0 + wn * 64 + tn * 16 + l16;
      float bv = bias[n];
#pragma unroll
      for (int r = 0; r < 8; r++) {
        int m = m0 + wm * 32 + tm * 16 + r + half * 8;
        float v = acc[tm][tn][r] + bv;
        if (MODE == 1) {
          Cout[(size_t)m * Ndim + n] = v;
        } else {
          int which = n >> 11;                 // 0=q 1=k 2=v
          int d  = n & (DMODEL - 1);
          int h  = d >> 7, hd = d & (HDIM - 1);
          int b2 = m >> 11, s = m & (S_LEN - 1);
          int bh = b2 * NHEAD + h;
          if (which == 0)
            Qo[((size_t)bh * S_LEN + s) * HDIM + hd] = (_Float16)(v * QSCALE_L2E);
          else if (which == 1)
            Ko[((size_t)bh * S_LEN + s) * HDIM + hd] = (_Float16)v;
          else
            Vto[((size_t)bh * HDIM + hd) * S_LEN + s] = (_Float16)v;
        }
      }
    }
  }
}

// ---------------------------------------------------------------------------
// Flash attention: one wave per 16-row Q tile, 32 keys per iteration.
// Causal mask analytic. Base-2 online softmax; row-max reduced across the
// 16-lane halves each iteration, but the row-SUM is kept as per-lane partials
// (alpha is half-uniform) and reduced once at the end — halves DS permute
// traffic inside the hot loop.
// ---------------------------------------------------------------------------
__global__ __launch_bounds__(256) void attn_fwd(
    const _Float16* __restrict__ Q, const _Float16* __restrict__ K,
    const _Float16* __restrict__ Vt, _Float16* __restrict__ AO) {
  __shared__ __attribute__((aligned(16))) _Float16 Pt[8][16 * LSTR];

  const int tid  = threadIdx.x;
  const int lane = tid & 31;
  const int wave = tid >> 5;
  const int l16  = lane & 15;
  const int half = lane >> 4;

  const int tile = blockIdx.x * 8 + wave;   // 4096 q-tiles total
  const int bh   = tile >> 7;               // /(S/16)
  const int q0   = (tile & 127) << 4;

  const _Float16* Qb = Q  + (size_t)bh * S_LEN * HDIM;
  const _Float16* Kb = K  + (size_t)bh * S_LEN * HDIM;
  const _Float16* Vb = Vt + (size_t)bh * HDIM * S_LEN;

  // Q tile fragments: 4 chunks of K=32 over HD=128 (scale+log2e pre-folded).
  v16h qf[4];
#pragma unroll
  for (int c = 0; c < 4; c++)
    qf[c] = frag16(Qb + (size_t)(q0 + l16) * HDIM + c * 32, half);

  f32x8 o[8] = {};
  float mrow[8], lrow[8];       // lrow: per-LANE partial sums
#pragma unroll
  for (int r = 0; r < 8; r++) { mrow[r] = -3.0e38f; lrow[r] = 0.0f; }

  const int ktend = (q0 + 16 + 31) >> 5;    // key tiles needed (causal)
  for (int kt = 0; kt < ktend; kt++) {
    const int kb = kt * 32;

    // scores: two 16x16 tiles covering 32 keys
    f32x8 s[2] = {};
#pragma unroll
    for (int st = 0; st < 2; st++) {
      const _Float16* kp = Kb + (size_t)(kb + st * 16 + l16) * HDIM;
#pragma unroll
      for (int c = 0; c < 4; c++)
        s[st] = wmma16(qf[c], frag16(kp + c * 32, half), s[st]);
    }

    // causal mask (only needed near the diagonal)
    if (kb + 31 > q0) {
#pragma unroll
      for (int st = 0; st < 2; st++) {
        int kk = kb + st * 16 + l16;
#pragma unroll
        for (int r = 0; r < 8; r++) {
          int qq = q0 + r + 8 * half;
          if (kk > qq) s[st][r] = -3.0e38f;
        }
      }
    }

    // online softmax (base 2); only the MAX needs a cross-lane reduction here
#pragma unroll
    for (int r = 0; r < 8; r++) {
      float v = fmaxf(s[0][r], s[1][r]);
#pragma unroll
      for (int off = 8; off >= 1; off >>= 1) v = fmaxf(v, __shfl_xor(v, off, 32));
      float mn    = fmaxf(mrow[r], v);
      float alpha = exp2f(mrow[r] - mn);     // uniform across each 16-lane half
      mrow[r]     = mn;
      float p0 = exp2f(s[0][r] - mn);
      float p1 = exp2f(s[1][r] - mn);
      s[0][r] = p0; s[1][r] = p1;
      lrow[r] = lrow[r] * alpha + p0 + p1;   // per-lane partial, no reduction
#pragma unroll
      for (int j = 0; j < 8; j++) o[j][r] *= alpha;
    }

    // P: C-layout -> A-layout transpose through per-wave LDS (in-order DS,
    // same-wave only -> no block barrier; trip counts differ per wave).
    _Float16* pt = &Pt[wave][0];
#pragma unroll
    for (int st = 0; st < 2; st++)
#pragma unroll
      for (int r = 0; r < 8; r++)
        pt[(r + 8 * half) * LSTR + st * 16 + l16] = (_Float16)s[st][r];
    v16h pf = frag16(pt + l16 * LSTR, half);

    // O += P * V  (V^T layout makes B-fragments contiguous along keys)
#pragma unroll
    for (int j = 0; j < 8; j++) {
      const _Float16* vp = Vb + (size_t)(j * 16 + l16) * S_LEN + kb;
      o[j] = wmma16(pf, frag16(vp, half), o[j]);
    }
  }

  // final cross-lane sum of the per-lane partials, then normalize + write
  const int b = bh >> 4, h = bh & (NHEAD - 1);
#pragma unroll
  for (int r = 0; r < 8; r++) {
    float ls = lrow[r];
#pragma unroll
    for (int off = 8; off >= 1; off >>= 1) ls += __shfl_xor(ls, off, 32);
    float inv = 1.0f / ls;
    size_t rowg = (size_t)b * S_LEN + q0 + r + 8 * half;
#pragma unroll
    for (int j = 0; j < 8; j++) {
      int col = h * HDIM + j * 16 + l16;
      AO[rowg * DMODEL + col] = (_Float16)(o[j][r] * inv);
    }
  }
}

// ---------------------------------------------------------------------------
// launch
// ---------------------------------------------------------------------------
extern "C" void kernel_launch(void* const* d_in, const int* in_sizes, int n_in,
                              void* d_out, int out_size, void* d_ws, size_t ws_size,
                              hipStream_t stream) {
  const float* x     = (const float*)d_in[0];
  const float* Wqkv  = (const float*)d_in[1];
  const float* bqkv  = (const float*)d_in[2];
  const float* Wproj = (const float*)d_in[3];
  const float* bproj = (const float*)d_in[4];
  // d_in[5] (mask) unused: causal structure handled analytically.
  float* out = (float*)d_out;

  _Float16* w = (_Float16*)d_ws;
  size_t off = 0;
  _Float16* xb     = w + off; off += (size_t)BS * S_LEN * DMODEL;      // 8.4M
  _Float16* wqkvt  = w + off; off += (size_t)3 * DMODEL * DMODEL;      // 12.6M
  _Float16* wprojt = w + off; off += (size_t)DMODEL * DMODEL;          // 4.2M
  _Float16* Qb     = w + off; off += (size_t)BS * S_LEN * DMODEL;
  _Float16* Kb     = w + off; off += (size_t)BS * S_LEN * DMODEL;
  _Float16* Vtb    = w + off; off += (size_t)BS * S_LEN * DMODEL;
  _Float16* AO     = w + off;                                          // total ~117MB

  const size_t nx = (size_t)BS * S_LEN * DMODEL;   // 8388608
  cast_f32_f16<<<nx / 1024, 256, 0, stream>>>(x, xb);
  transpose_cast<<<dim3(3 * DMODEL / 32, DMODEL / 32), 256, 0, stream>>>(
      Wqkv, wqkvt, DMODEL, 3 * DMODEL);
  transpose_cast<<<dim3(DMODEL / 32, DMODEL / 32), 256, 0, stream>>>(
      Wproj, wprojt, DMODEL, DMODEL);

  // QKV: [4096 x 6144] = xb[4096 x 2048] * Wqkv
  gemm_f16<0><<<dim3(3 * DMODEL / 128, BS * S_LEN / 128), 256, 0, stream>>>(
      xb, wqkvt, bqkv, nullptr, Qb, Kb, Vtb, DMODEL, 3 * DMODEL);

  // attention: 4096 q-tiles / 8 waves per block
  attn_fwd<<<512, 256, 0, stream>>>(Qb, Kb, Vtb, AO);

  // projection: out[4096 x 2048] fp32
  gemm_f16<1><<<dim3(DMODEL / 128, BS * S_LEN / 128), 256, 0, stream>>>(
      AO, wprojt, bproj, out, nullptr, nullptr, nullptr, DMODEL, DMODEL);
}
